// Transformer_10926396801513
// MI455X (gfx1250) — compile-verified, matrix operand order checked by reading, not run on hardware
//
#include <hip/hip_runtime.h>
#include <math.h>

// ---------------------------------------------------------------------------
// GPT forward for gfx1250 (MI455X), bf16 WMMA (v_wmma_f32_16x16x32_bf16).
// - All WMMA operands pre-converted to bf16 once (O(MK+KN)); weights packed
//   into per-lane fragment layout -> K-loop is pure b128 loads + WMMA.
// - GEMM register-blocked 32x32 per wave: 2 A-frags x 2 B-frags -> 4 WMMAs
//   per K-step, 2 loads/WMMA (halves VMEM issue pressure per matrix op).
// - V written transposed-bf16 so attention P*V fragments are b128 loads.
// Input order assumes setup_inputs() dict-insertion flattening:
//  0:x 1:w_embed 2:ln1_g 3:ln1_b 4:ln2_g 5:ln2_b 6:wq 7:bq 8:wk 9:bk 10:wv
// 11:bv 12:wo 13:bo 14:w1 15:b1 16:w2 17:b2 18:lnf_g 19:lnf_b 20:w_head
// ---------------------------------------------------------------------------

#define B_    8
#define T_    1024
#define E_    768
#define H4_   3072
#define NH_   12
#define HD_   64
#define L_    8
#define M_    (B_ * T_)
#define INF_  36
#define OUTN_ 64

typedef __attribute__((ext_vector_type(8)))  float  v8f;
typedef __attribute__((ext_vector_type(16))) __bf16 v16bf;
typedef unsigned short u16;
typedef unsigned int   u32;

union Frag { u32 u[8]; uint4 q[2]; v16bf v; };

__device__ __forceinline__ u16 f2bf(float f) {
  u32 u = __float_as_uint(f);
  u += 0x7FFFu + ((u >> 16) & 1u);              // RNE
  return (u16)(u >> 16);
}
__device__ __forceinline__ u32 pack2(float lo, float hi) {
  return (u32)f2bf(lo) | ((u32)f2bf(hi) << 16);
}
// ISA 7.12.2: 16-bit A/B^T fragment K-index for pair p, lane-half kh (0|8)
__device__ __forceinline__ int kmap(int p, int kh) {
  return ((p >> 2) << 4) + kh + ((p & 3) << 1);
}

// flags for k_gemm epilogue
#define GF_GELU   1   // exact-erf GELU before store
#define GF_BF16   2   // store bf16 row-major [M,N]
#define GF_BF16T  4   // store bf16 transposed [N,M]

// ------------------------------- embed -------------------------------------
__global__ void k_embed(const float* __restrict__ x, const float* __restrict__ we,
                        float* __restrict__ h) {
  int idx = blockIdx.x * 256 + threadIdx.x;
  if (idx >= M_ * E_) return;
  int e = idx % E_;
  int m = idx / E_;
  int t = m % T_;
  const float* xr = x + (size_t)m * INF_;
  float acc = 0.f;
#pragma unroll 4
  for (int f = 0; f < INF_; ++f) acc += xr[f] * we[f * E_ + e];
  float loc = -1.f + 2.f * (float)t / (float)(T_ - 1);
  acc += loc * we[INF_ * E_ + e];
  h[idx] = acc;
}

// --------------------------- weight fragment pack --------------------------
// fp32 W[K,N] -> bf16 pairs in per-lane WMMA fragment layout:
// Wp[(kt*(N/16)+nt)*256 + lane*8 + p] = {W[k][n], W[k+1][n]},
//   k = kt*32 + kmap(p, (lane>=16)*8), n = nt*16 + lane%16.
__global__ void k_packw(const float* __restrict__ W, u32* __restrict__ Wp,
                        int K, int N) {
  int i = blockIdx.x * 256 + threadIdx.x;
  if (i >= (K >> 1) * N) return;
  int p    = i & 7;
  int lane = (i >> 3) & 31;
  int tile = i >> 8;
  int ntiles = N >> 4;
  int nt = tile % ntiles;
  int kt = tile / ntiles;
  int k = kt * 32 + kmap(p, ((lane >> 4) << 3));
  int n = nt * 16 + (lane & 15);
  Wp[i] = pack2(W[(size_t)k * N + n], W[(size_t)(k + 1) * N + n]);
}

// ----------------------------- layernorm (fp32 -> bf16) --------------------
__global__ void __launch_bounds__(256) k_ln(const float* __restrict__ src,
                                            const float* __restrict__ g,
                                            const float* __restrict__ bta,
                                            u16* __restrict__ dst) {
  const int m   = blockIdx.x;
  const int tid = threadIdx.x;
  const float* row = src + (size_t)m * E_;
  float x0 = row[tid], x1 = row[tid + 256], x2 = row[tid + 512];

  __shared__ float sm[8], sv[8];
  float s = x0 + x1 + x2;
#pragma unroll
  for (int off = 16; off; off >>= 1) s += __shfl_xor(s, off, 32);
  if ((tid & 31) == 0) sm[tid >> 5] = s;
  __syncthreads();
  float tot = 0.f;
#pragma unroll
  for (int i = 0; i < 8; ++i) tot += sm[i];
  float mean = tot * (1.f / E_);

  float d0 = x0 - mean, d1 = x1 - mean, d2 = x2 - mean;
  float q = d0 * d0 + d1 * d1 + d2 * d2;
#pragma unroll
  for (int off = 16; off; off >>= 1) q += __shfl_xor(q, off, 32);
  if ((tid & 31) == 0) sv[tid >> 5] = q;
  __syncthreads();
  float vt = 0.f;
#pragma unroll
  for (int i = 0; i < 8; ++i) vt += sv[i];
  float rstd = rsqrtf(vt * (1.f / E_) + 1e-5f);

  u16* drow = dst + (size_t)m * E_;
  drow[tid]       = f2bf(d0 * rstd * g[tid]       + bta[tid]);
  drow[tid + 256] = f2bf(d1 * rstd * g[tid + 256] + bta[tid + 256]);
  drow[tid + 512] = f2bf(d2 * rstd * g[tid + 512] + bta[tid + 512]);
}

// ------------------------------- GEMM --------------------------------------
// D[M,N] = act(Abf16[M,K] @ Wpacked[K,N] + bias) (+ fp32 residual).
// M%128==0, N%64==0, K%32==0. 8 waves/block (4m x 2n), 32x32 tile per wave:
// 2 A-frags x 2 B-frags -> 4 WMMAs per K-step (2 b128 loads per WMMA).
__global__ void __launch_bounds__(256) k_gemm(const u16* __restrict__ Abf,
                                              const u32* __restrict__ Wp,
                                              const float* __restrict__ bias,
                                              const float* __restrict__ res,
                                              void* __restrict__ D,
                                              int M, int N, int K, int flags) {
  const int lane = threadIdx.x & 31;
  const int wave = threadIdx.x >> 5;
  const int m0 = blockIdx.y * 128 + (wave & 3) * 32;
  const int n0 = blockIdx.x * 64 + (wave >> 2) * 32;
  const int lm = lane & 15;
  const int kh = (lane >> 4) << 3;

  v8f acc[2][2] = {};
  const u16* arow0 = Abf + (size_t)(m0 + lm) * K + kh;
  const u16* arow1 = arow0 + (size_t)16 * K;
  const u32* wbase = Wp + (size_t)(n0 >> 4) * 256 + lane * 8;
  const size_t wkstride = (size_t)(N >> 4) * 256;   // u32 per k-tile

  for (int k0 = 0; k0 < K; k0 += 32) {
    __builtin_prefetch(arow0 + k0 + 32, 0, 1);
    __builtin_prefetch(arow1 + k0 + 32, 0, 1);
    __builtin_prefetch(wbase + ((size_t)(k0 >> 5) + 1) * wkstride, 0, 1);
    Frag a0, a1, b0, b1;
    a0.q[0] = *(const uint4*)(arow0 + k0);
    a0.q[1] = *(const uint4*)(arow0 + k0 + 16);
    a1.q[0] = *(const uint4*)(arow1 + k0);
    a1.q[1] = *(const uint4*)(arow1 + k0 + 16);
    const u32* wk = wbase + (size_t)(k0 >> 5) * wkstride;
    b0.q[0] = *(const uint4*)(wk);
    b0.q[1] = *(const uint4*)(wk + 4);
    b1.q[0] = *(const uint4*)(wk + 256);       // next 16-col tile
    b1.q[1] = *(const uint4*)(wk + 260);
    acc[0][0] = __builtin_amdgcn_wmma_f32_16x16x32_bf16(false, a0.v, false, b0.v, (short)0, acc[0][0], false, false);
    acc[0][1] = __builtin_amdgcn_wmma_f32_16x16x32_bf16(false, a0.v, false, b1.v, (short)0, acc[0][1], false, false);
    acc[1][0] = __builtin_amdgcn_wmma_f32_16x16x32_bf16(false, a1.v, false, b0.v, (short)0, acc[1][0], false, false);
    acc[1][1] = __builtin_amdgcn_wmma_f32_16x16x32_bf16(false, a1.v, false, b1.v, (short)0, acc[1][1], false, false);
  }

#pragma unroll
  for (int mt = 0; mt < 2; ++mt) {
#pragma unroll
    for (int nt = 0; nt < 2; ++nt) {
      const int rbase = m0 + mt * 16 + kh;     // C layout: rows r+kh
      const int col   = n0 + nt * 16 + lm;
      const float bvv = bias ? bias[col] : 0.f;
      float vals[8];
#pragma unroll
      for (int r = 0; r < 8; ++r) {
        float v = acc[mt][nt][r] + bvv;
        if (flags & GF_GELU) v = 0.5f * v * (1.f + erff(v * 0.70710678f));
        vals[r] = v;
      }
      if (flags & GF_BF16T) {                  // transposed bf16: Dt[col*M + row]
        uint4 pk;
        pk.x = pack2(vals[0], vals[1]);
        pk.y = pack2(vals[2], vals[3]);
        pk.z = pack2(vals[4], vals[5]);
        pk.w = pack2(vals[6], vals[7]);
        *(uint4*)((u16*)D + (size_t)col * M + rbase) = pk;
      } else if (flags & GF_BF16) {            // row-major bf16
        u16* Db = (u16*)D;
#pragma unroll
        for (int r = 0; r < 8; ++r) Db[(size_t)(rbase + r) * N + col] = f2bf(vals[r]);
      } else {                                 // fp32 (+ residual)
        float* Df = (float*)D;
#pragma unroll
        for (int r = 0; r < 8; ++r) {
          size_t idx = (size_t)(rbase + r) * N + col;
          float v = vals[r];
          if (res) v += res[idx];
          Df[idx] = v;
        }
      }
    }
  }
}

// --------------------------- flash attention -------------------------------
// One wave per (b, h, 16-query tile). Q,K bf16 row-major; V bf16 transposed
// (vT[e*M + token]); output Y bf16 row-major. Online softmax, 32 keys/step.
__global__ void __launch_bounds__(256) k_attn(const u16* __restrict__ Q,
                                              const u16* __restrict__ Kp,
                                              const u16* __restrict__ Vt,
                                              u16* __restrict__ Y) {
  __shared__ __align__(16) u16 Pst[8][16 * 32];
  const int lane = threadIdx.x & 31;
  const int wave = threadIdx.x >> 5;
  const int wid  = blockIdx.x * 8 + wave;
  const int qt = wid & 63;
  const int h  = (wid >> 6) % NH_;
  const int b  = wid / (64 * NH_);
  const int lm = lane & 15;
  const int kh = (lane >> 4) << 3;

  const size_t tokbase = (size_t)b * T_;
  const u16* qb = Q  + tokbase * E_ + h * HD_;
  const u16* kb = Kp + tokbase * E_ + h * HD_;

  Frag aq[2];
  {
    const u16* qrow = qb + (size_t)(qt * 16 + lm) * E_ + kh;
#pragma unroll
    for (int f = 0; f < 2; ++f) {
      aq[f].q[0] = *(const uint4*)(qrow + f * 32);
      aq[f].q[1] = *(const uint4*)(qrow + f * 32 + 16);
    }
  }

  float mi[8], li[8];
#pragma unroll
  for (int r = 0; r < 8; ++r) { mi[r] = -3.0e38f; li[r] = 0.f; }
  v8f o[4] = {};

  const int qhi = qt * 16 + 15;
  for (int j0 = 0; j0 <= qhi; j0 += 32) {
    Frag bk[2][2];
#pragma unroll
    for (int nt = 0; nt < 2; ++nt) {
      const u16* krow = kb + (size_t)(j0 + nt * 16 + lm) * E_ + kh;
#pragma unroll
      for (int kc = 0; kc < 2; ++kc) {
        bk[nt][kc].q[0] = *(const uint4*)(krow + kc * 32);
        bk[nt][kc].q[1] = *(const uint4*)(krow + kc * 32 + 16);
      }
    }
    v8f s0 = {}, s1 = {};
    s0 = __builtin_amdgcn_wmma_f32_16x16x32_bf16(false, aq[0].v, false, bk[0][0].v, (short)0, s0, false, false);
    s0 = __builtin_amdgcn_wmma_f32_16x16x32_bf16(false, aq[1].v, false, bk[0][1].v, (short)0, s0, false, false);
    s1 = __builtin_amdgcn_wmma_f32_16x16x32_bf16(false, aq[0].v, false, bk[1][0].v, (short)0, s1, false, false);
    s1 = __builtin_amdgcn_wmma_f32_16x16x32_bf16(false, aq[1].v, false, bk[1][1].v, (short)0, s1, false, false);

    const float sc = 0.125f;                 // 1/sqrt(64)
    const int key0 = j0 + lm, key1 = j0 + 16 + lm;
#pragma unroll
    for (int r = 0; r < 8; ++r) {
      const int row = qt * 16 + kh + r;
      float a0 = s0[r] * sc; if (key0 > row) a0 = -3.0e38f;
      float a1 = s1[r] * sc; if (key1 > row) a1 = -3.0e38f;

      float mx = fmaxf(a0, a1);
#pragma unroll
      for (int off = 1; off < 16; off <<= 1) mx = fmaxf(mx, __shfl_xor(mx, off, 32));
      const float nm = fmaxf(mi[r], mx);
      const float al = __expf(mi[r] - nm);
      const float e0 = __expf(a0 - nm);
      const float e1 = __expf(a1 - nm);
      float rs = e0 + e1;
#pragma unroll
      for (int off = 1; off < 16; off <<= 1) rs += __shfl_xor(rs, off, 32);
      li[r] = li[r] * al + rs;
      mi[r] = nm;
#pragma unroll
      for (int ct = 0; ct < 4; ++ct) o[ct][r] *= al;

      const int rowl = kh + r;
      Pst[wave][rowl * 32 + lm]      = f2bf(e0);
      Pst[wave][rowl * 32 + 16 + lm] = f2bf(e1);
    }
    asm volatile("s_wait_dscnt 0x0" ::: "memory");

    Frag pa;
#pragma unroll
    for (int p = 0; p < 8; ++p)
      pa.u[p] = *(const u32*)&Pst[wave][lm * 32 + kmap(p, kh)];

#pragma unroll
    for (int ct = 0; ct < 4; ++ct) {
      // V^T: contraction (keys) contiguous per lane-column -> b128 loads
      const u16* vcol = Vt + (size_t)(h * HD_ + ct * 16 + lm) * M_ + tokbase + j0 + kh;
      Frag bv;
      bv.q[0] = *(const uint4*)(vcol);
      bv.q[1] = *(const uint4*)(vcol + 16);
      o[ct] = __builtin_amdgcn_wmma_f32_16x16x32_bf16(false, pa.v, false, bv.v,
                                                      (short)0, o[ct], false, false);
    }
  }

#pragma unroll
  for (int ct = 0; ct < 4; ++ct)
#pragma unroll
    for (int r = 0; r < 8; ++r) {
      int row = qt * 16 + kh + r;
      Y[(tokbase + row) * E_ + h * HD_ + ct * 16 + lm] = f2bf(o[ct][r] / li[r]);
    }
}

// ------------------------------- host --------------------------------------
static inline void pgemm(const u16* A, const float* W, u32* wpack,
                         const float* bias, const float* res, void* D,
                         int N, int K, int flags, hipStream_t s) {
  int nu32 = (K >> 1) * N;
  k_packw<<<(nu32 + 255) / 256, 256, 0, s>>>(W, wpack, K, N);
  dim3 g(N / 64, M_ / 128);
  k_gemm<<<g, 256, 0, s>>>(A, wpack, bias, res, D, M_, N, K, flags);
}

extern "C" void kernel_launch(void* const* d_in, const int* in_sizes, int n_in,
                              void* d_out, int out_size, void* d_ws, size_t ws_size,
                              hipStream_t stream) {
  (void)in_sizes; (void)n_in; (void)out_size; (void)ws_size;
  const float* x       = (const float*)d_in[0];
  const float* w_embed = (const float*)d_in[1];
  const float* ln1_g   = (const float*)d_in[2];
  const float* ln1_b   = (const float*)d_in[3];
  const float* ln2_g   = (const float*)d_in[4];
  const float* ln2_b   = (const float*)d_in[5];
  const float* wq = (const float*)d_in[6];  const float* bq = (const float*)d_in[7];
  const float* wk = (const float*)d_in[8];  const float* bk = (const float*)d_in[9];
  const float* wv = (const float*)d_in[10]; const float* bv = (const float*)d_in[11];
  const float* wo = (const float*)d_in[12]; const float* bo = (const float*)d_in[13];
  const float* w1 = (const float*)d_in[14]; const float* b1 = (const float*)d_in[15];
  const float* w2 = (const float*)d_in[16]; const float* b2 = (const float*)d_in[17];
  const float* lnf_g  = (const float*)d_in[18];
  const float* lnf_b  = (const float*)d_in[19];
  const float* w_head = (const float*)d_in[20];

  const size_t ME = (size_t)M_ * E_;
  float* h0   = (float*)d_ws;               // fp32 residual stream (ping)
  float* h1   = h0 + ME;                    // fp32 residual stream (pong)
  u16*  lnb   = (u16*)(h1 + ME);            // bf16 LN output (GEMM A)
  u16*  qb    = lnb + ME;                   // bf16 Q  [m, h*64+d]
  u16*  kbuf  = qb + ME;                    // bf16 K
  u16*  vT    = kbuf + ME;                  // bf16 V transposed [e, m]
  u16*  yb    = vT + ME;                    // bf16 attention out
  u16*  mid   = yb + ME;                    // bf16 MLP mid [m, 4E]
  u32*  wpack = (u32*)(mid + (size_t)M_ * H4_);  // packed weight fragments

  k_embed<<<(M_ * E_ + 255) / 256, 256, 0, stream>>>(x, w_embed, h0);

  float* hc = h0;
  float* hn = h1;
  for (int l = 0; l < L_; ++l) {
    const size_t oE = (size_t)l * E_, oEE = (size_t)l * E_ * E_;
    const size_t oH = (size_t)l * H4_, oEH = (size_t)l * E_ * H4_;

    k_ln<<<M_, 256, 0, stream>>>(hc, ln1_g + oE, ln1_b + oE, lnb);
    pgemm(lnb, wq + oEE, wpack, bq + oE, nullptr, qb,   E_, E_, GF_BF16,  stream);
    pgemm(lnb, wk + oEE, wpack, bk + oE, nullptr, kbuf, E_, E_, GF_BF16,  stream);
    pgemm(lnb, wv + oEE, wpack, bv + oE, nullptr, vT,   E_, E_, GF_BF16T, stream);

    k_attn<<<B_ * NH_ * (T_ / 16) / 8, 256, 0, stream>>>(qb, kbuf, vT, yb);

    pgemm(yb, wo + oEE, wpack, bo + oE, hc, hn, E_, E_, 0, stream);           // h += y@wo+bo
    k_ln<<<M_, 256, 0, stream>>>(hn, ln2_g + oE, ln2_b + oE, lnb);
    pgemm(lnb, w1 + oEH, wpack, b1 + oH, nullptr, mid, H4_, E_,
          GF_GELU | GF_BF16, stream);                                         // gelu -> bf16
    pgemm(mid, w2 + oEH, wpack, b2 + oE, hn, hc, E_, H4_, 0, stream);         // h += mid@w2+b2
  }

  k_ln<<<M_, 256, 0, stream>>>(hc, lnf_g, lnf_b, lnb);
  pgemm(lnb, w_head, wpack, nullptr, nullptr, (float*)d_out, OUTN_, E_, 0, stream);
}
